// DFusion_7730941132895
// MI455X (gfx1250) — matrix operand order repaired and checked by reading, not compiled, not run on hardware
//
#include <hip/hip_runtime.h>
#include <hip/hip_bf16.h>

typedef __bf16 bf16_t;
typedef __attribute__((ext_vector_type(16))) __bf16 v16bf;
typedef __attribute__((ext_vector_type(8)))  __bf16 v8bf;
typedef __attribute__((ext_vector_type(8)))  float  v8f;
typedef unsigned int u32x4_t __attribute__((ext_vector_type(4)));
typedef unsigned int u32x8_t __attribute__((ext_vector_type(8)));

// Problem constants (fixed by the reference)
constexpr int Bsz = 8, Hn = 16, Lseq = 384, Dm = 256, Mm = 3, Nn = 128;
constexpr int NBLK  = Bsz * Hn * Mm * Mm;    // 1152 softmax/attention blocks
constexpr int ROWS  = NBLK * Nn;             // 147456 rows of x
constexpr int ROWS4 = Bsz * Hn * Mm * Nn;    // 49152 rows of xm / u

// ---------------------------------------------------------------------------
// CDNA5 async / TDM helpers (CDNA5 ISA)
// ---------------------------------------------------------------------------
__device__ inline unsigned lds_offset_of(const void* p) {
  // generic LDS address: low 32 bits are the workgroup-relative LDS offset
  return (unsigned)(unsigned long long)p;
}

__device__ inline void async_ldg_b128(unsigned lds_off, const void* g) {
  asm volatile("global_load_async_to_lds_b128 %0, %1, off"
               :: "v"(lds_off), "v"((unsigned long long)g) : "memory");
}

__device__ inline void wait_async0() {
  asm volatile("s_wait_asynccnt 0x0" ::: "memory");
}

__device__ inline void wait_tensor0() {
  asm volatile("s_wait_tensorcnt 0x0" ::: "memory");
}

// One-shot TDM load of a contiguous run of `nelem` bf16 elements into LDS.
// D# per ISA §8.3/8.4: group0 = {flags, lds_addr, global_addr, type=2},
// group1 = {data_size=2B, tensor_dim0=nelem, tile_dim0=nelem, tile_dim1=1,
//           tensor_dim0_stride=nelem}.  Groups 2/3 = null (<=2D tensor).
__device__ inline void tdm_load_1d(unsigned lds_off, const void* gptr,
                                   unsigned nelem /* bf16 elems, <= 65535 */) {
  unsigned long long ga = (unsigned long long)gptr;
  u32x4_t g0;
  g0.x = 1u;                                                // count=1 user D#
  g0.y = lds_off;                                           // lds_addr
  g0.z = (unsigned)ga;                                      // global_addr[31:0]
  g0.w = (unsigned)((ga >> 32) & 0x01ffffffu) | (2u << 30); // addr[56:32]|type=2
  u32x8_t g1;
  g1.s0 = 1u << 16;                   // workgroup_mask=0, data_size=1 (2 bytes)
  g1.s1 = (nelem & 0xffffu) << 16;    // tensor_dim0[15:0]
  g1.s2 = (nelem >> 16) & 0xffffu;    // tensor_dim0[31:16], tensor_dim1=0
  g1.s3 = (nelem & 0xffffu) << 16;    // tile_dim0
  g1.s4 = 1u;                         // tile_dim1=1, tile_dim2=0
  g1.s5 = nelem;                      // tensor_dim0_stride[31:0]
  g1.s6 = 0u;
  g1.s7 = 0u;
  asm volatile("tensor_load_to_lds %0, %1, null, null"
               :: "s"(g0), "s"(g1) : "memory");
}

// ---------------------------------------------------------------------------
// Fragment helpers (layouts per CDNA5 ISA §7.12.2, wave32)
// ---------------------------------------------------------------------------
__device__ inline v8f vzero8() {
  v8f z = {0.f, 0.f, 0.f, 0.f, 0.f, 0.f, 0.f, 0.f};
  return z;
}

// A-matrix 16x32 bf16, row-major source with leading dim `ld` (elements).
// lanes 0-15: row=lane,   K {k0..k0+7} in v0-3, K {k0+16..k0+23} in v4-7
// lanes16-31: row=lane-16,K {k0+8..k0+15},      K {k0+24..k0+31}
__device__ inline v16bf load_frag_a(const bf16_t* base, int ld, int lane,
                                    int row0, int k0) {
  int r  = row0 + (lane & 15);
  int ks = k0 + ((lane >> 4) << 3);
  const bf16_t* p = base + (size_t)r * ld + ks;
  v8bf lo = *reinterpret_cast<const v8bf*>(p);
  v8bf hi = *reinterpret_cast<const v8bf*>(p + 16);
  return __builtin_shufflevector(lo, hi, 0, 1, 2, 3, 4, 5, 6, 7,
                                 8, 9, 10, 11, 12, 13, 14, 15);
}

// B-matrix 32x16 bf16 loaded from B-transposed storage (N rows x K cols,
// leading dim `ld`).  lane<16: col=n0+lane holds K {k0..k0+7, k0+16..k0+23};
// lane>=16: col=n0+lane-16 holds K {k0+8..k0+15, k0+24..k0+31}.
__device__ inline v16bf load_frag_bt(const bf16_t* bt, int ld, int lane,
                                     int n0, int k0) {
  int c  = n0 + (lane & 15);
  int ks = k0 + ((lane >> 4) << 3);
  const v8bf* p = reinterpret_cast<const v8bf*>(bt + (size_t)c * ld + ks);
  v8bf lo = p[0];
  v8bf hi = p[2];
  return __builtin_shufflevector(lo, hi, 0, 1, 2, 3, 4, 5, 6, 7,
                                 8, 9, 10, 11, 12, 13, 14, 15);
}

__device__ inline v8f wmma_bf16(v16bf a, v16bf b, v8f c) {
  return __builtin_amdgcn_wmma_f32_16x16x32_bf16(false, a, false, b,
                                                 (short)0, c, false, false);
}

// ---------------------------------------------------------------------------
// 1) f32 -> bf16 convert (weights)
// ---------------------------------------------------------------------------
__global__ void cvt_bf16_kernel(const float* __restrict__ in,
                                bf16_t* __restrict__ out, int n) {
  int idx = blockIdx.x * 256 + threadIdx.x;
  if (idx < n) out[idx] = (bf16_t)in[idx];
}

// ---------------------------------------------------------------------------
// 2) value -> bf16, transposed per (b,h,mk) block: Vt[bhm][dd][j]
// ---------------------------------------------------------------------------
__global__ void vtrans_kernel(const float* __restrict__ V,
                              bf16_t* __restrict__ Vt) {
  size_t idx = (size_t)blockIdx.x * 256 + threadIdx.x;   // < 12,582,912
  size_t bhm = idx / (size_t)(Dm * Nn);
  int rem = (int)(idx % (size_t)(Dm * Nn));
  int dd = rem >> 7;
  int j  = rem & 127;
  size_t bh = bhm / Mm;
  int mk = (int)(bhm % Mm);
  Vt[idx] = (bf16_t)V[(bh * Lseq + (size_t)mk * Nn + j) * Dm + dd];
}

// ---------------------------------------------------------------------------
// 3) block softmax: one wave per 128-wide key-block row -> bf16 subscores
//    subscores layout: [bb][i][j] with bb = ((b*H+h)*M+mq)*M+mk
// ---------------------------------------------------------------------------
__global__ void __launch_bounds__(128) softmax_kernel(
    const float* __restrict__ Attn, bf16_t* __restrict__ Sout) {
  int rid  = blockIdx.x * 4 + (threadIdx.x >> 5);  // 0..147455
  int lane = threadIdx.x & 31;
  int i  = rid & 127;
  int bb = rid >> 7;
  int mk = bb % 3;
  int mq = (bb / 3) % 3;
  int bh = bb / 9;
  const float* row =
      Attn + ((size_t)bh * Lseq + (size_t)mq * Nn + i) * Lseq + (size_t)mk * Nn;
  float v[4];
  float m = -3.4e38f;
#pragma unroll
  for (int t = 0; t < 4; ++t) {
    v[t] = row[lane + 32 * t];
    m = fmaxf(m, v[t]);
  }
#pragma unroll
  for (int off = 16; off; off >>= 1) m = fmaxf(m, __shfl_xor(m, off, 32));
  float s = 0.f;
#pragma unroll
  for (int t = 0; t < 4; ++t) {
    v[t] = __expf(v[t] - m);
    s += v[t];
  }
#pragma unroll
  for (int off = 16; off; off >>= 1) s += __shfl_xor(s, off, 32);
  float inv = 1.0f / s;
  bf16_t* orow = Sout + (size_t)rid * Nn;
#pragma unroll
  for (int t = 0; t < 4; ++t) orow[lane + 32 * t] = (bf16_t)(v[t] * inv);
}

// ---------------------------------------------------------------------------
// 4) GEMM1: per block bb, X[128x256] = S[128x128] @ V[128x256], bf16 out.
//    A tile staged to LDS (async); B fragments software-pipelined.
// ---------------------------------------------------------------------------
__global__ void __launch_bounds__(128) gemm1_kernel(
    const bf16_t* __restrict__ S, const bf16_t* __restrict__ Vt,
    bf16_t* __restrict__ X) {
  __shared__ __align__(16) bf16_t sA[16 * Nn];   // 4 KB A tile
  int wg = blockIdx.x;           // 9216
  int bb = wg >> 3;
  int rt = wg & 7;
  int lane = threadIdx.x & 31;
  int wave = threadIdx.x >> 5;
  int bh = bb / 9;
  int mk = bb % 3;
  const bf16_t* Sblk = S + (size_t)bb * (Nn * Nn);
  const bf16_t* Vblk = Vt + ((size_t)bh * Mm + mk) * (Dm * Nn);
  int m0 = rt * 16;

  // stage the 16x128 A tile (contiguous 4KB) into LDS asynchronously
  {
    const char* gsrc = (const char*)(Sblk + (size_t)m0 * Nn);
    unsigned base = lds_offset_of(sA);
#pragma unroll
    for (int c = 0; c < 2; ++c) {
      int chunk = threadIdx.x + c * 128;       // 256 x 16B
      async_ldg_b128(base + chunk * 16, gsrc + chunk * 16);
    }
    wait_async0();
  }
  __syncthreads();

  v8f acc[4];
#pragma unroll
  for (int t = 0; t < 4; ++t) acc[t] = vzero8();

  // software-pipelined K loop: prefetch step k+1 before the WMMAs of step k
  v16bf a = load_frag_a(sA, Nn, lane, 0, 0);
  v16bf bcur[4], bnxt[4];
#pragma unroll
  for (int t = 0; t < 4; ++t)
    bcur[t] = load_frag_bt(Vblk, Nn, lane, (wave * 4 + t) * 16, 0);
#pragma unroll
  for (int kk = 0; kk < Nn; kk += 32) {
    int kn = kk + 32;
    v16bf an = a;
    if (kn < Nn) {
      an = load_frag_a(sA, Nn, lane, 0, kn);
#pragma unroll
      for (int t = 0; t < 4; ++t)
        bnxt[t] = load_frag_bt(Vblk, Nn, lane, (wave * 4 + t) * 16, kn);
    }
#pragma unroll
    for (int t = 0; t < 4; ++t) acc[t] = wmma_bf16(a, bcur[t], acc[t]);
    a = an;
#pragma unroll
    for (int t = 0; t < 4; ++t) bcur[t] = bnxt[t];
  }

  int col = lane & 15;
  int rb  = (lane >> 4) * 8;
  size_t rowbase = (size_t)bb * Nn + m0;
#pragma unroll
  for (int t = 0; t < 4; ++t) {
    int n0 = (wave * 4 + t) * 16;
#pragma unroll
    for (int r = 0; r < 8; ++r)
      X[(rowbase + rb + r) * Dm + n0 + col] = (bf16_t)acc[t][r];
  }
}

// ---------------------------------------------------------------------------
// 5) mean over the key-modality axis (v8bf vectorized)
// ---------------------------------------------------------------------------
__global__ void mean_kernel(const bf16_t* __restrict__ X,
                            bf16_t* __restrict__ Xm) {
  size_t vidx = (size_t)blockIdx.x * 256 + threadIdx.x;  // ROWS4*32 chunks
  size_t ru = vidx >> 5;
  int c8 = (int)(vidx & 31);
  size_t G = ru >> 7;
  int i = (int)(ru & 127);
  const v8bf* p = reinterpret_cast<const v8bf*>(X + (G * 384 + i) * Dm) + c8;
  v8bf a = p[0];
  v8bf b = p[(size_t)128 * Dm / 8];
  v8bf c = p[(size_t)256 * Dm / 8];
  v8bf r;
#pragma unroll
  for (int e = 0; e < 8; ++e)
    r[e] = (bf16_t)(((float)a[e] + (float)b[e] + (float)c[e]) * (1.0f / 3.0f));
  reinterpret_cast<v8bf*>(Xm)[vidx] = r;
}

// ---------------------------------------------------------------------------
// 6) u = xm @ Lw^T (f32 out).  A tile staged to LDS via one TDM descriptor;
//    B fragments software-pipelined.
// ---------------------------------------------------------------------------
__global__ void __launch_bounds__(128) gemmU_kernel(
    const bf16_t* __restrict__ Xm, const bf16_t* __restrict__ Lw,
    float* __restrict__ U) {
  __shared__ __align__(16) bf16_t sA[16 * Dm];   // 8 KB A tile
  int r0 = blockIdx.x * 16;      // 3072 WGs
  int lane = threadIdx.x & 31;
  int wave = threadIdx.x >> 5;

  if (threadIdx.x < 32) {        // wave 0 issues the TDM op (EXEC ignored)
    tdm_load_1d(lds_offset_of(sA), Xm + (size_t)r0 * Dm, 16 * Dm);
    wait_tensor0();
  }
  __syncthreads();

  v8f acc[4];
#pragma unroll
  for (int t = 0; t < 4; ++t) acc[t] = vzero8();

  v16bf a = load_frag_a(sA, Dm, lane, 0, 0);
  v16bf bcur[4], bnxt[4];
#pragma unroll
  for (int t = 0; t < 4; ++t)
    bcur[t] = load_frag_bt(Lw, Dm, lane, (wave * 4 + t) * 16, 0);
#pragma unroll
  for (int kk = 0; kk < Dm; kk += 32) {
    int kn = kk + 32;
    v16bf an = a;
    if (kn < Dm) {
      an = load_frag_a(sA, Dm, lane, 0, kn);
#pragma unroll
      for (int t = 0; t < 4; ++t)
        bnxt[t] = load_frag_bt(Lw, Dm, lane, (wave * 4 + t) * 16, kn);
    }
#pragma unroll
    for (int t = 0; t < 4; ++t) acc[t] = wmma_bf16(a, bcur[t], acc[t]);
    a = an;
#pragma unroll
    for (int t = 0; t < 4; ++t) bcur[t] = bnxt[t];
  }

  int col = lane & 15;
  int rb  = (lane >> 4) * 8;
#pragma unroll
  for (int t = 0; t < 4; ++t) {
    int n0 = (wave * 4 + t) * 16;
#pragma unroll
    for (int r = 0; r < 8; ++r)
      U[(size_t)(r0 + rb + r) * Dm + n0 + col] = acc[t][r];
  }
}

// ---------------------------------------------------------------------------
// 7) layer GEMM: y = x@Gw^T + Gb - u ; then LN+ELU (->bf16) or final scatter
//    to d_out (f32).  A tile staged via async copy; B pipelined; LN in LDS.
// ---------------------------------------------------------------------------
__global__ void __launch_bounds__(128) gemm_layer_kernel(
    const bf16_t* __restrict__ Xin, const bf16_t* __restrict__ W,
    const float* __restrict__ Gb, const float* __restrict__ U,
    const float* __restrict__ gamma, const float* __restrict__ beta,
    bf16_t* __restrict__ Xout, float* __restrict__ Out, int is_final) {
  __shared__ __align__(16) bf16_t sA[16 * Dm];   // 8 KB A tile
  __shared__ float tile[16][256];
  __shared__ float ps[16][8];
  __shared__ float pq[16][8];
  __shared__ float mu_s[16];
  __shared__ float rs_s[16];

  int r0 = blockIdx.x * 16;      // 9216 WGs
  int lane = threadIdx.x & 31;
  int wave = threadIdx.x >> 5;

  {
    const char* gsrc = (const char*)(Xin + (size_t)r0 * Dm);
    unsigned base = lds_offset_of(sA);
#pragma unroll
    for (int c = 0; c < 4; ++c) {
      int chunk = threadIdx.x + c * 128;       // 512 x 16B = 8 KB
      async_ldg_b128(base + chunk * 16, gsrc + chunk * 16);
    }
    wait_async0();
  }
  __syncthreads();

  v8f acc[4];
#pragma unroll
  for (int t = 0; t < 4; ++t) acc[t] = vzero8();

  v16bf a = load_frag_a(sA, Dm, lane, 0, 0);
  v16bf bcur[4], bnxt[4];
#pragma unroll
  for (int t = 0; t < 4; ++t)
    bcur[t] = load_frag_bt(W, Dm, lane, (wave * 4 + t) * 16, 0);
#pragma unroll
  for (int kk = 0; kk < Dm; kk += 32) {
    int kn = kk + 32;
    v16bf an = a;
    if (kn < Dm) {
      an = load_frag_a(sA, Dm, lane, 0, kn);
#pragma unroll
      for (int t = 0; t < 4; ++t)
        bnxt[t] = load_frag_bt(W, Dm, lane, (wave * 4 + t) * 16, kn);
    }
#pragma unroll
    for (int t = 0; t < 4; ++t) acc[t] = wmma_bf16(a, bcur[t], acc[t]);
    a = an;
#pragma unroll
    for (int t = 0; t < 4; ++t) bcur[t] = bnxt[t];
  }

  int col16 = lane & 15;
  int rb    = (lane >> 4) * 8;

  if (is_final) {
#pragma unroll
    for (int t = 0; t < 4; ++t) {
      int c = (wave * 4 + t) * 16 + col16;
#pragma unroll
      for (int r = 0; r < 8; ++r) {
        int gr = r0 + rb + r;
        int i  = gr & 127;
        int G  = gr / 384;               // (b*H+h)*M + mq
        int ru = G * 128 + i;
        float v = acc[t][r] + Gb[c] - U[(size_t)ru * Dm + c];
        // torch-style reshape/permute scatter
        int b_  = gr / 18432;            // M*M*n*H rows per batch
        int h_  = (gr / 1152) & 15;      // M*M*n rows per head
        int mq  = (gr / 384) % 3;
        int mkk = (gr >> 7) % 3;
        int q = mq * 384 + mkk * 128 + i;
        int l = q / 3;
        int s = q - l * 3;
        Out[(((size_t)b_ * Lseq + l) * Hn + h_) * 768 + (size_t)s * Dm + c] = v;
      }
    }
  } else {
#pragma unroll
    for (int t = 0; t < 4; ++t) {
      int c = (wave * 4 + t) * 16 + col16;
#pragma unroll
      for (int r = 0; r < 8; ++r) {
        int gr = r0 + rb + r;
        int ru = (gr / 384) * 128 + (gr & 127);
        tile[rb + r][c] = acc[t][r] + Gb[c] - U[(size_t)ru * Dm + c];
      }
    }
    __syncthreads();
    // per-row LayerNorm reduction: 8 threads per row, 32 cols each
    int row = threadIdx.x >> 3;
    int seg = threadIdx.x & 7;
    float s = 0.f, sq = 0.f;
#pragma unroll
    for (int j = 0; j < 32; ++j) {
      float x = tile[row][seg * 32 + j];
      s += x;
      sq += x * x;
    }
    ps[row][seg] = s;
    pq[row][seg] = sq;
    __syncthreads();
    if (threadIdx.x < 16) {
      float S = 0.f, Q = 0.f;
#pragma unroll
      for (int j = 0; j < 8; ++j) {
        S += ps[threadIdx.x][j];
        Q += pq[threadIdx.x][j];
      }
      float mu  = S * (1.0f / 256.0f);
      float var = Q * (1.0f / 256.0f) - mu * mu;
      mu_s[threadIdx.x] = mu;
      rs_s[threadIdx.x] = rsqrtf(var + 1e-5f);
    }
    __syncthreads();
    float mu = mu_s[row];
    float rs = rs_s[row];
    size_t gr = (size_t)(r0 + row);
#pragma unroll
    for (int j = 0; j < 32; ++j) {
      int c = seg * 32 + j;
      float x = (tile[row][c] - mu) * rs * gamma[c] + beta[c];
      x = x > 0.f ? x : (__expf(x) - 1.0f);   // ELU (alpha=1)
      Xout[gr * Dm + c] = (bf16_t)x;
    }
  }
}

// ---------------------------------------------------------------------------
// Host side
// ---------------------------------------------------------------------------
extern "C" void kernel_launch(void* const* d_in, const int* in_sizes, int n_in,
                              void* d_out, int out_size, void* d_ws,
                              size_t ws_size, hipStream_t stream) {
  (void)in_sizes; (void)n_in; (void)out_size; (void)ws_size;
  const float* value    = (const float*)d_in[0];
  const float* attn     = (const float*)d_in[1];
  // d_in[2] = scores: unused by the reference forward
  const float* Gw       = (const float*)d_in[3];
  const float* Gb       = (const float*)d_in[4];
  const float* Lw       = (const float*)d_in[5];
  const float* ln_gamma = (const float*)d_in[6];
  const float* ln_beta  = (const float*)d_in[7];
  float* out = (float*)d_out;

  // workspace layout (bytes, all 256-aligned)
  char* ws = (char*)d_ws;
  constexpr size_t SUB_B = (size_t)NBLK * Nn * Nn * 2;
  constexpr size_t VT_B  = (size_t)Bsz * Hn * Mm * Dm * Nn * 2;
  constexpr size_t W_B   = (size_t)3 * Dm * Dm * 2;
  constexpr size_t X_B   = (size_t)ROWS * Dm * 2;
  constexpr size_t XM_B  = (size_t)ROWS4 * Dm * 2;
  size_t off = 0;
  bf16_t* sub = (bf16_t*)(ws + off); off += SUB_B;
  bf16_t* vt  = (bf16_t*)(ws + off); off += VT_B;
  bf16_t* gwb = (bf16_t*)(ws + off); off += W_B;
  bf16_t* lwb = (bf16_t*)(ws + off); off += W_B;
  bf16_t* xA  = (bf16_t*)(ws + off); off += X_B;
  bf16_t* xB  = (bf16_t*)(ws + off); off += X_B;
  bf16_t* xm  = (bf16_t*)(ws + off); off += XM_B;
  float*  u   = (float*)(ws + off);  off += (size_t)ROWS4 * Dm * 4;

  const int wN = 3 * Dm * Dm;  // 196608
  cvt_bf16_kernel<<<(wN + 255) / 256, 256, 0, stream>>>(Gw, gwb, wN);
  cvt_bf16_kernel<<<(wN + 255) / 256, 256, 0, stream>>>(Lw, lwb, wN);
  vtrans_kernel<<<(int)((size_t)Bsz * Hn * Mm * Dm * Nn / 256), 256, 0,
                  stream>>>(value, vt);
  softmax_kernel<<<ROWS / 4, 128, 0, stream>>>(attn, sub);
  gemm1_kernel<<<NBLK * 8, 128, 0, stream>>>(sub, vt, xA);

  bf16_t* xcur = xA;
  bf16_t* xnxt = xB;
  for (int l = 0; l < 3; ++l) {
    mean_kernel<<<ROWS4 * (Dm / 8) / 256, 256, 0, stream>>>(xcur, xm);
    gemmU_kernel<<<ROWS4 / 16, 128, 0, stream>>>(xm, lwb + (size_t)l * Dm * Dm,
                                                 u);
    int is_final = (l == 2) ? 1 : 0;
    const float* g = ln_gamma + (l < 2 ? l : 0) * Dm;
    const float* bta = ln_beta + (l < 2 ? l : 0) * Dm;
    gemm_layer_kernel<<<ROWS / 16, 128, 0, stream>>>(
        xcur, gwb + (size_t)l * Dm * Dm, Gb + (size_t)l * Dm, u, g, bta, xnxt,
        out, is_final);
    bf16_t* tmp = xcur; xcur = xnxt; xnxt = tmp;
  }
}
